// Sae_37847251812602
// MI455X (gfx1250) — compile-verified, matrix operand order checked by reading, not run on hardware
//
#include <hip/hip_runtime.h>
#include <stdint.h>

// ---------------------------------------------------------------------------
// Problem dims (fixed by the reference)
// ---------------------------------------------------------------------------
#define D_IN    1280
#define N_LAT   20480
#define N_TOK   4096
#define TOTAL_PRE ((size_t)N_TOK * (size_t)N_LAT)   // 83,886,080
#define TOTAL_K   (32u * (unsigned)N_TOK)           // 131,072 (global top-k)

// ---------------------------------------------------------------------------
// Trivially-constructible vector types (safe inside unions)
// ---------------------------------------------------------------------------
typedef __attribute__((ext_vector_type(16))) __bf16   bf16x16;
typedef __attribute__((ext_vector_type(8)))  float    f32x8;
typedef __attribute__((ext_vector_type(4)))  float    f32x4;
typedef __attribute__((ext_vector_type(4)))  unsigned u32x4;

union FragB16 { bf16x16 v; u32x4 q[2]; };

// float -> bf16 bits, round-to-nearest-even (manual; avoids scalar __bf16 cvt)
static __device__ __forceinline__ unsigned short f2bf_bits(float f) {
  unsigned u = __float_as_uint(f);
  unsigned rnd = 0x7FFFu + ((u >> 16) & 1u);
  return (unsigned short)((u + rnd) >> 16);
}

// ---------------------------------------------------------------------------
// Workspace layout (bytes). Zeroed via hipMemsetAsync each launch.
// ---------------------------------------------------------------------------
#define WS_HIST1   0u                      // 4096  x u32 (coarse 12-bit hist)
#define WS_HIST2   16384u                  // 2^19  x u32 (fine 19-bit hist)
#define WS_PARAMS  (16384u + 2097152u)     // u32: [0]=B1 [1]=need1 [2]=T [3]=ties [4]=l0cnt [5]=tiectr
#define WS_SCAL    (WS_PARAMS + 32u)       // f32: [0]=l2_total [1]=tv_total
#define WS_CMEAN   (WS_SCAL + 8u)          // 1280 x f32
#define WS_PTL2    (WS_CMEAN + 5120u)      // 4096 x f32
#define WS_PTTV    (WS_PTL2 + 16384u)      // 4096 x f32
#define WS_END     (WS_PTTV + 16384u)      // ~2.15 MB total

// ===========================================================================
// 1) Encoder GEMM:  pre = relu((x - b_dec) @ W_enc^T + b_enc)   [WMMA bf16]
//    Tile: 128(M) x 128(N), BK=32.  8 waves -> each wave 32x64 (2x4 tiles).
// ===========================================================================
__launch_bounds__(256)
__global__ void sae_encode_wmma(const float* __restrict__ x,
                                const float* __restrict__ W_enc,
                                const float* __restrict__ b_enc,
                                const float* __restrict__ b_dec,
                                float* __restrict__ pre,
                                unsigned* __restrict__ l0_ctr) {
  __shared__ __align__(16) unsigned short lA[128][32];   // bf16 bits, 8 KB
  __shared__ __align__(16) unsigned short lB[128][32];   // bf16 bits, 8 KB

  const int tid  = threadIdx.x;
  const int lane = tid & 31;
  const int w    = tid >> 5;
  const int wmi  = w & 3;                 // 4 wave-rows  (32 M each)
  const int wni  = w >> 2;                // 2 wave-cols  (64 N each)
  const int m0   = blockIdx.x * 128;
  const int n0   = blockIdx.y * 128;

  const int ldr  = tid >> 1;              // 0..127 staging row
  const int ldc  = (tid & 1) * 16;        // 0 / 16 staging col base

  f32x8 acc[2][4];
#pragma unroll
  for (int a = 0; a < 2; ++a)
#pragma unroll
    for (int b = 0; b < 4; ++b)
#pragma unroll
      for (int i = 0; i < 8; ++i) acc[a][b][i] = 0.f;

  const float* __restrict__ xrow = x     + (size_t)(m0 + ldr) * D_IN;
  const float* __restrict__ wrow = W_enc + (size_t)(n0 + ldr) * D_IN;

  for (int k0 = 0; k0 < D_IN; k0 += 32) {
    // ---- stage tiles to LDS as bf16 (16 elements per thread per tile) ----
    {
      const f32x4* sa = (const f32x4*)(xrow  + k0 + ldc);
      const f32x4* sb = (const f32x4*)(wrow  + k0 + ldc);
      const f32x4* bd = (const f32x4*)(b_dec + k0 + ldc);
#pragma unroll
      for (int q = 0; q < 4; ++q) {
        f32x4 a4 = sa[q], d4 = bd[q], b4 = sb[q];
#pragma unroll
        for (int e = 0; e < 4; ++e) {
          lA[ldr][ldc + q * 4 + e] = f2bf_bits(a4[e] - d4[e]);
          lB[ldr][ldc + q * 4 + e] = f2bf_bits(b4[e]);
        }
      }
      if (k0 + 32 < D_IN) {               // gfx1250 global_prefetch of next tiles
        __builtin_prefetch(wrow + k0 + 32 + ldc, 0, 1);
        __builtin_prefetch(xrow + k0 + 32 + ldc, 0, 1);
      }
    }
    __syncthreads();

    // ---- build fragments: lane<16 holds K 0-7 / 16-23, lane>=16 K 8-15 / 24-31
    const int kh = (lane >> 4) * 8;
    FragB16 fa[2], fb[4];
#pragma unroll
    for (int mt = 0; mt < 2; ++mt) {
      int r = wmi * 32 + mt * 16 + (lane & 15);
      fa[mt].q[0] = *(const u32x4*)&lA[r][kh];
      fa[mt].q[1] = *(const u32x4*)&lA[r][kh + 16];
    }
#pragma unroll
    for (int nt = 0; nt < 4; ++nt) {
      int c = wni * 64 + nt * 16 + (lane & 15);
      fb[nt].q[0] = *(const u32x4*)&lB[c][kh];
      fb[nt].q[1] = *(const u32x4*)&lB[c][kh + 16];
    }
#pragma unroll
    for (int mt = 0; mt < 2; ++mt)
#pragma unroll
      for (int nt = 0; nt < 4; ++nt)
        acc[mt][nt] = __builtin_amdgcn_wmma_f32_16x16x32_bf16(
            false, fa[mt].v, false, fb[nt].v, (short)0, acc[mt][nt], false, false);
    __syncthreads();
  }

  // ---- epilogue: +b_enc, ReLU, store, l0 count ----
  int cnt = 0;
#pragma unroll
  for (int mt = 0; mt < 2; ++mt) {
    int mb = m0 + wmi * 32 + mt * 16 + ((lane >> 4) * 8);   // VGPR i -> row mb+i
#pragma unroll
    for (int nt = 0; nt < 4; ++nt) {
      int nb = n0 + wni * 64 + nt * 16 + (lane & 15);
      float be = b_enc[nb];
#pragma unroll
      for (int i = 0; i < 8; ++i) {
        float v = acc[mt][nt][i] + be;
        v = v > 0.f ? v : 0.f;
        cnt += (v > 0.f) ? 1 : 0;
        pre[(size_t)(mb + i) * N_LAT + nb] = v;
      }
    }
  }
  for (int off = 16; off; off >>= 1) cnt += __shfl_xor(cnt, off, 32);
  if (lane == 0) atomicAdd(l0_ctr, (unsigned)cnt);
}

// ===========================================================================
// 2a) Coarse histogram over bits[31:19] of positive activations (LDS-private)
// ===========================================================================
__launch_bounds__(256)
__global__ void hist1_kernel(const float* __restrict__ pre, unsigned* __restrict__ hist) {
  __shared__ unsigned lh[4096];
  for (int i = threadIdx.x; i < 4096; i += 256) lh[i] = 0;
  __syncthreads();
  size_t stride = (size_t)gridDim.x * 256 * 4;
  for (size_t i = ((size_t)blockIdx.x * 256 + threadIdx.x) * 4; i < TOTAL_PRE; i += stride) {
    f32x4 v = *(const f32x4*)(pre + i);
#pragma unroll
    for (int e = 0; e < 4; ++e)
      if (v[e] > 0.f) atomicAdd(&lh[__float_as_uint(v[e]) >> 19], 1u);
  }
  __syncthreads();
  for (int i = threadIdx.x; i < 4096; i += 256)
    if (lh[i]) atomicAdd(&hist[i], lh[i]);
}

// 2b) Find coarse bucket B1 and residual count need1 (scan from the top)
__launch_bounds__(256)
__global__ void find1_kernel(const unsigned* __restrict__ hist, unsigned* __restrict__ params) {
  __shared__ unsigned csum[256];
  const int t = threadIdx.x;
  unsigned s = 0;
  for (int i = 0; i < 16; ++i) s += hist[t * 16 + i];
  csum[t] = s;
  __syncthreads();
  if (t == 0) {
    unsigned long long cum = 0;
    unsigned B1 = 0, need = TOTAL_K;
    for (int c = 255; c >= 0; --c) {
      if (cum + csum[c] >= (unsigned long long)TOTAL_K) {
        for (int b = c * 16 + 15; b >= c * 16; --b) {
          unsigned h = hist[b];
          if (cum + h >= (unsigned long long)TOTAL_K) {
            B1 = (unsigned)b; need = (unsigned)(TOTAL_K - cum); goto done1;
          }
          cum += h;
        }
      }
      cum += csum[c];
    }
done1:
    params[0] = B1;
    params[1] = need;
  }
}

// 2c) Fine histogram of low 19 bits inside bucket B1 (resolves exact threshold)
__launch_bounds__(256)
__global__ void hist2_kernel(const float* __restrict__ pre,
                             const unsigned* __restrict__ params,
                             unsigned* __restrict__ hist2) {
  const unsigned B1 = params[0];
  size_t stride = (size_t)gridDim.x * 256 * 4;
  for (size_t i = ((size_t)blockIdx.x * 256 + threadIdx.x) * 4; i < TOTAL_PRE; i += stride) {
    f32x4 v = *(const f32x4*)(pre + i);
#pragma unroll
    for (int e = 0; e < 4; ++e) {
      if (v[e] > 0.f) {
        unsigned u = __float_as_uint(v[e]);
        if ((u >> 19) == B1) atomicAdd(&hist2[u & 0x7FFFFu], 1u);
      }
    }
  }
}

// 2d) Find exact bit threshold T and tie budget
__launch_bounds__(256)
__global__ void find2_kernel(const unsigned* __restrict__ hist2, unsigned* __restrict__ params) {
  __shared__ unsigned csum[256];
  const int t = threadIdx.x;
  const unsigned B1 = params[0], need = params[1];
  unsigned s = 0;
  for (int i = 0; i < 2048; ++i) s += hist2[t * 2048 + i];
  csum[t] = s;
  __syncthreads();
  if (t == 0) {
    unsigned long long cum = 0;
    unsigned T = (B1 << 19), ties = need;
    for (int c = 255; c >= 0; --c) {
      if (cum + csum[c] >= (unsigned long long)need) {
        for (int b = c * 2048 + 2047; b >= c * 2048; --b) {
          unsigned h = hist2[b];
          if (cum + h >= (unsigned long long)need) {
            T = (B1 << 19) | (unsigned)b; ties = (unsigned)(need - cum); goto done2;
          }
          cum += h;
        }
      }
      cum += csum[c];
    }
done2:
    params[2] = T;
    params[3] = ties;
  }
}

// 3) Zero everything below threshold (ties kept up to the exact global budget)
__launch_bounds__(256)
__global__ void sparsify_kernel(float* __restrict__ pre,
                                const unsigned* __restrict__ params,
                                unsigned* __restrict__ tie_ctr) {
  const unsigned T = params[2], ties = params[3];
  size_t stride = (size_t)gridDim.x * 256 * 4;
  for (size_t i = ((size_t)blockIdx.x * 256 + threadIdx.x) * 4; i < TOTAL_PRE; i += stride) {
    f32x4 v = *(const f32x4*)(pre + i);
#pragma unroll
    for (int e = 0; e < 4; ++e) {
      float f = v[e];
      bool keep = false;
      if (f > 0.f) {
        unsigned u = __float_as_uint(f);
        if (u > T) keep = true;
        else if (u == T) keep = (atomicAdd(tie_ctr, 1u) < ties);
      }
      if (!keep) v[e] = 0.f;
    }
    *(f32x4*)(pre + i) = v;
  }
}

// ===========================================================================
// 4) Sparse decode: one workgroup per token.
//    sae_out = top_acts @ W_dec + b_dec; fused per-token L2 + top indices.
// ===========================================================================
__launch_bounds__(256)
__global__ void decode_kernel(const float* __restrict__ top_acts,
                              const float* __restrict__ W_dec,
                              const float* __restrict__ b_dec,
                              const float* __restrict__ x,
                              float* __restrict__ sae_out,
                              float* __restrict__ top_idx_out,
                              float* __restrict__ ptl2,
                              float* __restrict__ l2_total) {
  const int t = blockIdx.x;
  const int tid = threadIdx.x;
  __shared__ int   s_cnt;
  __shared__ int   s_idx[256];
  __shared__ float s_val[256];
  __shared__ int   s_top[32];
  __shared__ int   s_ntop;
  __shared__ float s_red[256];

  if (tid == 0) s_ntop = 0;
  float acc[5] = {0.f, 0.f, 0.f, 0.f, 0.f};
  const float* row = top_acts + (size_t)t * N_LAT;

  for (int base = 0; base < N_LAT; base += 256) {
    if (tid == 0) s_cnt = 0;
    __syncthreads();
    float v = row[base + tid];
    if (v != 0.f) { int p = atomicAdd(&s_cnt, 1); s_idx[p] = base + tid; s_val[p] = v; }
    __syncthreads();
    const int c = s_cnt;
    for (int e = 0; e < c; ++e) {
      const float* wr = W_dec + (size_t)s_idx[e] * D_IN;
      float vv = s_val[e];
#pragma unroll
      for (int r = 0; r < 5; ++r) acc[r] += vv * wr[tid + r * 256];
    }
    if (tid == 0) {
      for (int e = 0; e < c && s_ntop < 32; ++e) s_top[s_ntop++] = s_idx[e];
    }
    __syncthreads();
  }

  float es = 0.f;
#pragma unroll
  for (int r = 0; r < 5; ++r) {
    int d = tid + r * 256;
    float o = acc[r] + b_dec[d];
    sae_out[(size_t)t * D_IN + d] = o;
    float e = o - x[(size_t)t * D_IN + d];
    es += e * e;
  }
  s_red[tid] = es;
  __syncthreads();
  for (int off = 128; off; off >>= 1) {
    if (tid < off) s_red[tid] += s_red[tid + off];
    __syncthreads();
  }
  if (tid == 0) { ptl2[t] = s_red[0]; atomicAdd(l2_total, s_red[0]); }
  if (tid < 32) top_idx_out[(size_t)t * 32 + tid] = (tid < s_ntop) ? (float)s_top[tid] : 0.f;
}

// ===========================================================================
// 5) Statistics
// ===========================================================================
__launch_bounds__(256)
__global__ void colmean_kernel(const float* __restrict__ x, float* __restrict__ cmean) {
  int d = blockIdx.x * 256 + threadIdx.x;
  if (d >= D_IN) return;
  float s = 0.f;
  for (int t = 0; t < N_TOK; ++t) s += x[(size_t)t * D_IN + d];
  cmean[d] = s * (1.f / (float)N_TOK);
}

__launch_bounds__(256)
__global__ void tv_kernel(const float* __restrict__ x, const float* __restrict__ cmean,
                          float* __restrict__ pttv, float* __restrict__ tv_total) {
  const int t = blockIdx.x, tid = threadIdx.x;
  __shared__ float red[256];
  float s = 0.f;
#pragma unroll
  for (int r = 0; r < 5; ++r) {
    int d = tid + r * 256;
    float c = x[(size_t)t * D_IN + d] - cmean[d];
    s += c * c;
  }
  red[tid] = s;
  __syncthreads();
  for (int off = 128; off; off >>= 1) {
    if (tid < off) red[tid] += red[tid + off];
    __syncthreads();
  }
  if (tid == 0) { pttv[t] = red[0]; atomicAdd(tv_total, red[0]); }
}

__launch_bounds__(256)
__global__ void finalize_kernel(const float* __restrict__ ptl2, const float* __restrict__ pttv,
                                const float* __restrict__ scal, const unsigned* __restrict__ params,
                                float* __restrict__ out_fvu, float* __restrict__ out_l0,
                                float* __restrict__ out_l2,  float* __restrict__ out_auxk,
                                float* __restrict__ out_ev) {
  int i = blockIdx.x * 256 + threadIdx.x;
  if (i < N_TOK) out_ev[i] = 1.f - ptl2[i] / pttv[i];
  if (i == 0) {
    float l2 = scal[0], tv = scal[1];
    out_fvu[0]  = l2 / tv;
    out_l0[0]   = (float)params[4] / (float)N_TOK;
    out_l2[0]   = l2 / (float)((size_t)N_TOK * (size_t)D_IN);
    out_auxk[0] = 0.f;
  }
}

// ===========================================================================
// Host launch
// ===========================================================================
extern "C" void kernel_launch(void* const* d_in, const int* in_sizes, int n_in,
                              void* d_out, int out_size, void* d_ws, size_t ws_size,
                              hipStream_t stream) {
  (void)in_sizes; (void)n_in; (void)out_size; (void)ws_size;
  const float* x     = (const float*)d_in[0];
  const float* W_enc = (const float*)d_in[1];
  const float* b_enc = (const float*)d_in[2];
  const float* W_dec = (const float*)d_in[3];
  const float* b_dec = (const float*)d_in[4];

  float* out      = (float*)d_out;
  float* out_sae  = out;                                   // [4096,1280]
  float* out_acts = out + (size_t)N_TOK * D_IN;            // [4096,20480]
  float* out_tidx = out_acts + TOTAL_PRE;                  // [4096,32]
  float* out_fvu  = out_tidx + (size_t)N_TOK * 32;
  float* out_l0   = out_fvu + 1;
  float* out_l2   = out_l0 + 1;
  float* out_auxk = out_l2 + 1;
  float* out_ev   = out_auxk + 1;                          // [4096]

  char* ws = (char*)d_ws;
  unsigned* hist1  = (unsigned*)(ws + WS_HIST1);
  unsigned* hist2  = (unsigned*)(ws + WS_HIST2);
  unsigned* params = (unsigned*)(ws + WS_PARAMS);
  float*    scal   = (float*)(ws + WS_SCAL);
  float*    cmean  = (float*)(ws + WS_CMEAN);
  float*    ptl2   = (float*)(ws + WS_PTL2);
  float*    pttv   = (float*)(ws + WS_PTTV);

  hipMemsetAsync(d_ws, 0, WS_END, stream);

  dim3 egrid(N_TOK / 128, N_LAT / 128);                    // 32 x 160
  sae_encode_wmma<<<egrid, 256, 0, stream>>>(x, W_enc, b_enc, b_dec, out_acts, &params[4]);

  hist1_kernel<<<2048, 256, 0, stream>>>(out_acts, hist1);
  find1_kernel<<<1, 256, 0, stream>>>(hist1, params);
  hist2_kernel<<<2048, 256, 0, stream>>>(out_acts, params, hist2);
  find2_kernel<<<1, 256, 0, stream>>>(hist2, params);
  sparsify_kernel<<<2048, 256, 0, stream>>>(out_acts, params, &params[5]);

  decode_kernel<<<N_TOK, 256, 0, stream>>>(out_acts, W_dec, b_dec, x,
                                           out_sae, out_tidx, ptl2, &scal[0]);
  colmean_kernel<<<(D_IN + 255) / 256, 256, 0, stream>>>(x, cmean);
  tv_kernel<<<N_TOK, 256, 0, stream>>>(x, cmean, pttv, &scal[1]);
  finalize_kernel<<<(N_TOK + 255) / 256, 256, 0, stream>>>(ptl2, pttv, scal, params,
                                                           out_fvu, out_l0, out_l2,
                                                           out_auxk, out_ev);
}
